// GlobalCrossAttention_17454747091611
// MI455X (gfx1250) — matrix operand order repaired and checked by reading, not compiled, not run on hardware
//
#include <hip/hip_runtime.h>
#include <hip/hip_fp16.h>

typedef _Float16 h8  __attribute__((ext_vector_type(8)));
typedef _Float16 h16 __attribute__((ext_vector_type(16)));
typedef float    f8  __attribute__((ext_vector_type(8)));

#define C_DIM   256
#define H_DIM   8
#define D_DIM   32
#define NQ      900
#define NQP     912     // 57 * 16, padded query rows
#define QTILES  57
#define NKV     4096
#define WGRID   64
#define BATCH   2
#define SCALE_Q 0.17677669529663687f   // 32^-0.5

// ---------------------------------------------------------------------------
// Generic projection: Y = X(f32) @ W^T(f32) + b, emitted as f16 into a
// layout chosen per consumer. One wave computes one 16x16 tile via 8 WMMAs.
// mode 0: Q -> Qh[b][h][q][d]   (scaled, rows >= Mvalid zero-filled)
// mode 1: K -> Kh[b][h][n][d]
// mode 2: V -> Vt[b][h][d][n]   (transposed for the PV B-fragment)
// ---------------------------------------------------------------------------
__global__ void __launch_bounds__(128) proj_kernel(
    const float* __restrict__ X, const float* __restrict__ W,
    const float* __restrict__ bias, _Float16* __restrict__ out,
    int mode, int Mvalid, int Mtiles, float scale)
{
    int wid   = blockIdx.x * 4 + (threadIdx.x >> 5);
    int total = BATCH * Mtiles * 16;
    if (wid >= total) return;
    int lane = threadIdx.x & 31, colL = lane & 15, hf = lane >> 4;
    int b    = wid / (Mtiles * 16);
    int rem  = wid % (Mtiles * 16);
    int m0   = (rem >> 4) * 16, n0 = (rem & 15) * 16;

    int arow = m0 + colL; if (arow >= Mvalid) arow = Mvalid - 1;
    const float* xrow = X + ((size_t)b * Mvalid + arow) * C_DIM;
    const float* wrow = W + (size_t)(n0 + colL) * C_DIM;

    f8 acc = {};
    #pragma unroll
    for (int ks = 0; ks < 8; ++ks) {
        int kb = ks * 32;
        float fa[16], fb[16];
        *(float4*)(fa + 0)  = *(const float4*)(xrow + kb + 16 * hf);
        *(float4*)(fa + 4)  = *(const float4*)(xrow + kb + 16 * hf + 4);
        *(float4*)(fa + 8)  = *(const float4*)(xrow + kb + 16 * hf + 8);
        *(float4*)(fa + 12) = *(const float4*)(xrow + kb + 16 * hf + 12);
        *(float4*)(fb + 0)  = *(const float4*)(wrow + kb + 16 * hf);
        *(float4*)(fb + 4)  = *(const float4*)(wrow + kb + 16 * hf + 4);
        *(float4*)(fb + 8)  = *(const float4*)(wrow + kb + 16 * hf + 8);
        *(float4*)(fb + 12) = *(const float4*)(wrow + kb + 16 * hf + 12);
        h16 a, bf;
        #pragma unroll
        for (int i = 0; i < 16; ++i) { a[i] = (_Float16)fa[i]; bf[i] = (_Float16)fb[i]; }
        acc = __builtin_amdgcn_wmma_f32_16x16x32_f16(false, a, false, bf,
                                                     (short)0, acc, false, false);
    }

    float bvv = bias[n0 + colL];
    #pragma unroll
    for (int r = 0; r < 8; ++r) {
        int m = m0 + r + 8 * hf;
        int c = n0 + colL;
        int hd = c >> 5, d = c & 31;
        float v = (acc[r] + bvv) * scale;
        if (mode == 0) {
            if (m >= Mvalid) v = 0.0f;
            out[((size_t)(b * H_DIM + hd) * NQP + m) * D_DIM + d] = (_Float16)v;
        } else if (mode == 1) {
            out[((size_t)(b * H_DIM + hd) * NKV + m) * D_DIM + d] = (_Float16)v;
        } else {
            out[((size_t)(b * H_DIM + hd) * D_DIM + d) * NKV + m] = (_Float16)v;
        }
    }
}

// ---------------------------------------------------------------------------
// RPE MLP: per (b,q,axis,coord) point, 2 -> 512 relu -> 8.
// Weights cached in LDS; one wave per point, lane covers 16 hidden units.
// Output layout [b][h][coord][NQP] so attention reads 8 contiguous q-rows.
// ---------------------------------------------------------------------------
__global__ void __launch_bounds__(256) rpe_kernel(
    const float* __restrict__ ref2d,
    const float* __restrict__ w1x, const float* __restrict__ b1x, const float* __restrict__ w2x,
    const float* __restrict__ w1y, const float* __restrict__ b1y, const float* __restrict__ w2y,
    float* __restrict__ rpeX, float* __restrict__ rpeY)
{
    __shared__ float lds[11264];
    float* Lw1x = lds;         float* Lb1x = lds + 1024; float* Lw2x = lds + 1536;
    float* Lw1y = lds + 5632;  float* Lb1y = lds + 6656; float* Lw2y = lds + 7168;
    for (int i = threadIdx.x; i < 1024; i += blockDim.x) { Lw1x[i] = w1x[i]; Lw1y[i] = w1y[i]; }
    for (int i = threadIdx.x; i < 512;  i += blockDim.x) { Lb1x[i] = b1x[i]; Lb1y[i] = b1y[i]; }
    for (int i = threadIdx.x; i < 4096; i += blockDim.x) { Lw2x[i] = w2x[i]; Lw2y[i] = w2y[i]; }
    __syncthreads();

    int lane  = threadIdx.x & 31;
    int wglob = blockIdx.x * 8 + (threadIdx.x >> 5);
    int nw    = gridDim.x * 8;
    const int total = BATCH * NQ * 2 * WGRID;

    for (int p = wglob; p < total; p += nw) {
        int b  = p / (NQ * 128);
        int r1 = p % (NQ * 128);
        int q  = r1 >> 7;
        int t  = r1 & 127;
        int axis = t >> 6;
        int u    = t & 63;

        const float* rr = ref2d + ((size_t)b * NQ + q) * 4;
        float lo, hi;
        if (axis == 0) { lo = rr[0] - 0.5f * rr[2]; hi = rr[0] + 0.5f * rr[2]; }
        else           { lo = rr[1] - 0.5f * rr[3]; hi = rr[1] + 0.5f * rr[3]; }
        float pos = ((float)u + 0.5f) * 16.0f;
        float d0 = lo - pos, d1 = hi - pos;

        const float* w1 = axis ? Lw1y : Lw1x;
        const float* b1 = axis ? Lb1y : Lb1x;
        const float* w2 = axis ? Lw2y : Lw2x;

        float acc[8];
        #pragma unroll
        for (int hh = 0; hh < 8; ++hh) acc[hh] = 0.0f;

        for (int j = 0; j < 16; ++j) {
            int r = lane + j * 32;
            float hv = fmaxf(w1[2 * r] * d0 + w1[2 * r + 1] * d1 + b1[r], 0.0f);
            #pragma unroll
            for (int hh = 0; hh < 8; ++hh) acc[hh] += hv * w2[hh * 512 + r];
        }
        #pragma unroll
        for (int hh = 0; hh < 8; ++hh)
            for (int mm = 1; mm < 32; mm <<= 1)
                acc[hh] += __shfl_xor(acc[hh], mm, 32);

        if (lane == 0) {
            float* outp = axis ? rpeY : rpeX;
            #pragma unroll
            for (int hh = 0; hh < 8; ++hh)
                outp[((size_t)(b * H_DIM + hh) * WGRID + u) * NQP + q] = acc[hh];
        }
    }
}

// ---------------------------------------------------------------------------
// Flash attention: one wave per (b, head, 16-query tile).
// Streams 32-key blocks: 2 S-WMMAs + rpe/mask + online softmax + 2 PV-WMMAs.
// P re-layout (C-frag -> A-frag) via a 1KB per-wave LDS round trip.
// ---------------------------------------------------------------------------
__global__ void __launch_bounds__(128) attn_kernel(
    const _Float16* __restrict__ Qh, const _Float16* __restrict__ Kh,
    const _Float16* __restrict__ Vt, const float* __restrict__ rpeX,
    const float* __restrict__ rpeY, const int* __restrict__ pmask,
    _Float16* __restrict__ Xh)
{
    __shared__ __align__(16) _Float16 Pbuf[4][16 * 32];
    int wl  = threadIdx.x >> 5;
    int wid = blockIdx.x * 4 + wl;
    if (wid >= BATCH * H_DIM * QTILES) return;
    int qt = wid % QTILES;
    int hd = (wid / QTILES) % H_DIM;
    int b  = wid / (QTILES * H_DIM);
    int lane = threadIdx.x & 31, colL = lane & 15, hf = lane >> 4;
    int q0 = qt * 16;
    _Float16* P = Pbuf[wl];
    size_t bh = (size_t)(b * H_DIM + hd);

    // Q A-fragment (16x32, row = lane, contiguous K: k = 16*hf + i)
    const _Float16* qrow = Qh + (bh * NQP + q0 + colL) * D_DIM;
    h16 aq;
    {
        h8 g0 = *(const h8*)(qrow + 16 * hf);
        h8 g1 = *(const h8*)(qrow + 16 * hf + 8);
        #pragma unroll
        for (int i = 0; i < 8; ++i) { aq[i] = g0[i]; aq[i + 8] = g1[i]; }
    }

    float m_run[8], l_run[8];
    #pragma unroll
    for (int r = 0; r < 8; ++r) { m_run[r] = -3.0e38f; l_run[r] = 0.0f; }
    f8 o0 = {}, o1 = {};

    const _Float16* kbase  = Kh   + bh * NKV * D_DIM;
    const _Float16* vbase  = Vt   + bh * D_DIM * NKV;
    const float*    rxbase = rpeX + bh * WGRID * NQP;
    const float*    rybase = rpeY + bh * WGRID * NQP;
    const int*      mbase  = pmask + b * NKV;

    for (int g = 0; g < NKV / 32; ++g) {
        float sv[2][8];
        #pragma unroll
        for (int c2 = 0; c2 < 2; ++c2) {
            int n0  = g * 32 + c2 * 16;
            int key = n0 + colL;
            h16 bk = *(const h16*)(kbase + (size_t)key * D_DIM + 16 * hf);
            f8 zc = {};
            f8 s = __builtin_amdgcn_wmma_f32_16x16x32_f16(false, aq, false, bk,
                                                          (short)0, zc, false, false);
            int x = (n0 & 63) + colL;
            int y = n0 >> 6;
            float rx[8], ry[8];
            *(float4*)(rx + 0) = *(const float4*)(rxbase + (size_t)x * NQP + q0 + 8 * hf);
            *(float4*)(rx + 4) = *(const float4*)(rxbase + (size_t)x * NQP + q0 + 8 * hf + 4);
            *(float4*)(ry + 0) = *(const float4*)(rybase + (size_t)y * NQP + q0 + 8 * hf);
            *(float4*)(ry + 4) = *(const float4*)(rybase + (size_t)y * NQP + q0 + 8 * hf + 4);
            float madd = -100.0f * (float)mbase[key];
            #pragma unroll
            for (int r = 0; r < 8; ++r) sv[c2][r] = s[r] + rx[r] + ry[r] + madd;
        }

        // online softmax over this 32-key block (rows live in VGPR index,
        // key columns in lanes -> intra-half shuffle reductions)
        float mx[8];
        #pragma unroll
        for (int r = 0; r < 8; ++r) mx[r] = fmaxf(sv[0][r], sv[1][r]);
        #pragma unroll
        for (int dd = 1; dd < 16; dd <<= 1) {
            #pragma unroll
            for (int r = 0; r < 8; ++r) mx[r] = fmaxf(mx[r], __shfl_xor(mx[r], dd, 32));
        }
        float sum[8];
        #pragma unroll
        for (int r = 0; r < 8; ++r) {
            float mnew = fmaxf(m_run[r], mx[r]);
            float rs   = __expf(m_run[r] - mnew);
            m_run[r]   = mnew;
            l_run[r]  *= rs;
            o0[r] *= rs; o1[r] *= rs;
            float p0 = __expf(sv[0][r] - mnew);
            float p1 = __expf(sv[1][r] - mnew);
            P[(r + 8 * hf) * 32 + colL]      = (_Float16)p0;
            P[(r + 8 * hf) * 32 + 16 + colL] = (_Float16)p1;
            sum[r] = p0 + p1;
        }
        #pragma unroll
        for (int dd = 1; dd < 16; dd <<= 1) {
            #pragma unroll
            for (int r = 0; r < 8; ++r) sum[r] += __shfl_xor(sum[r], dd, 32);
        }
        #pragma unroll
        for (int r = 0; r < 8; ++r) l_run[r] += sum[r];

        // P (16x32) back as A-fragment from LDS, then PV WMMAs
        h16 ap;
        {
            const _Float16* prow = P + colL * 32;
            h8 g0 = *(const h8*)(prow + 16 * hf);
            h8 g1 = *(const h8*)(prow + 16 * hf + 8);
            #pragma unroll
            for (int i = 0; i < 8; ++i) { ap[i] = g0[i]; ap[i + 8] = g1[i]; }
        }
        int kb2 = g * 32 + 16 * hf;
        h16 bv0 = *(const h16*)(vbase + (size_t)colL * NKV + kb2);
        h16 bv1 = *(const h16*)(vbase + (size_t)(colL + 16) * NKV + kb2);
        o0 = __builtin_amdgcn_wmma_f32_16x16x32_f16(false, ap, false, bv0,
                                                    (short)0, o0, false, false);
        o1 = __builtin_amdgcn_wmma_f32_16x16x32_f16(false, ap, false, bv1,
                                                    (short)0, o1, false, false);
    }

    #pragma unroll
    for (int r = 0; r < 8; ++r) {
        float inv = 1.0f / l_run[r];
        int qrowi = q0 + r + 8 * hf;
        _Float16* orow = Xh + ((size_t)b * NQP + qrowi) * C_DIM + hd * D_DIM;
        orow[colL]      = (_Float16)(o0[r] * inv);
        orow[colL + 16] = (_Float16)(o1[r] * inv);
    }
}

// ---------------------------------------------------------------------------
// Output projection: out = Xh(f16) @ Wp^T(f32) + bp, f32 result.
// ---------------------------------------------------------------------------
__global__ void __launch_bounds__(128) outproj_kernel(
    const _Float16* __restrict__ Xh, const float* __restrict__ Wp,
    const float* __restrict__ bp, float* __restrict__ out)
{
    int wid = blockIdx.x * 4 + (threadIdx.x >> 5);
    const int total = (BATCH * NQP / 16) * 16;   // 114 * 16 tiles
    if (wid >= total) return;
    int m0 = (wid >> 4) * 16, n0 = (wid & 15) * 16;
    int lane = threadIdx.x & 31, colL = lane & 15, hf = lane >> 4;

    const _Float16* xrow = Xh + (size_t)(m0 + colL) * C_DIM;
    const float*    wrow = Wp + (size_t)(n0 + colL) * C_DIM;

    f8 acc = {};
    #pragma unroll
    for (int ks = 0; ks < 8; ++ks) {
        int kb = ks * 32;
        h16 a;
        {
            h8 g0 = *(const h8*)(xrow + kb + 16 * hf);
            h8 g1 = *(const h8*)(xrow + kb + 16 * hf + 8);
            #pragma unroll
            for (int i = 0; i < 8; ++i) { a[i] = g0[i]; a[i + 8] = g1[i]; }
        }
        float fb[16];
        *(float4*)(fb + 0)  = *(const float4*)(wrow + kb + 16 * hf);
        *(float4*)(fb + 4)  = *(const float4*)(wrow + kb + 16 * hf + 4);
        *(float4*)(fb + 8)  = *(const float4*)(wrow + kb + 16 * hf + 8);
        *(float4*)(fb + 12) = *(const float4*)(wrow + kb + 16 * hf + 12);
        h16 bf;
        #pragma unroll
        for (int i = 0; i < 16; ++i) bf[i] = (_Float16)fb[i];
        acc = __builtin_amdgcn_wmma_f32_16x16x32_f16(false, a, false, bf,
                                                     (short)0, acc, false, false);
    }

    float bvv = bp[n0 + colL];
    #pragma unroll
    for (int r = 0; r < 8; ++r) {
        int mg = m0 + r + 8 * hf;
        int b = mg / NQP, q = mg % NQP;
        if (q < NQ)
            out[((size_t)b * NQ + q) * C_DIM + n0 + colL] = acc[r] + bvv;
    }
}

// ---------------------------------------------------------------------------
extern "C" void kernel_launch(void* const* d_in, const int* in_sizes, int n_in,
                              void* d_out, int out_size, void* d_ws, size_t ws_size,
                              hipStream_t stream)
{
    (void)in_sizes; (void)n_in; (void)out_size; (void)ws_size;
    const float* query = (const float*)d_in[0];
    const float* ref2d = (const float*)d_in[1];
    const float* kin   = (const float*)d_in[2];
    const float* vin   = (const float*)d_in[3];
    const int*   pmask = (const int*)d_in[4];
    const float* Wq  = (const float*)d_in[7];
    const float* bq  = (const float*)d_in[8];
    const float* Wk  = (const float*)d_in[9];
    const float* bk  = (const float*)d_in[10];
    const float* Wv  = (const float*)d_in[11];
    const float* bvp = (const float*)d_in[12];
    const float* Wp  = (const float*)d_in[13];
    const float* bp  = (const float*)d_in[14];
    const float* c1w1 = (const float*)d_in[15];
    const float* c1b1 = (const float*)d_in[16];
    const float* c1w2 = (const float*)d_in[17];
    const float* c2w1 = (const float*)d_in[18];
    const float* c2b1 = (const float*)d_in[19];
    const float* c2w2 = (const float*)d_in[20];

    char* ws = (char*)d_ws;
    size_t off = 0;
    auto wsalloc = [&](size_t bytes) {
        void* p = ws + off;
        off += (bytes + 255) & ~(size_t)255;
        return p;
    };
    _Float16* Qh   = (_Float16*)wsalloc((size_t)BATCH * H_DIM * NQP * D_DIM * 2);
    _Float16* Kh   = (_Float16*)wsalloc((size_t)BATCH * H_DIM * NKV * D_DIM * 2);
    _Float16* Vt   = (_Float16*)wsalloc((size_t)BATCH * H_DIM * D_DIM * NKV * 2);
    float*    rpeX = (float*)   wsalloc((size_t)BATCH * H_DIM * WGRID * NQP * 4);
    float*    rpeY = (float*)   wsalloc((size_t)BATCH * H_DIM * WGRID * NQP * 4);
    _Float16* Xh   = (_Float16*)wsalloc((size_t)BATCH * NQP * C_DIM * 2);

    // Projections (Q: 2*57*16 tiles / 4 waves per block = 456 blocks; K,V: 2048)
    proj_kernel<<<456,  128, 0, stream>>>(query, Wq, bq,  Qh, 0, NQ,  QTILES, SCALE_Q);
    proj_kernel<<<2048, 128, 0, stream>>>(kin,   Wk, bk,  Kh, 1, NKV, 256,    1.0f);
    proj_kernel<<<2048, 128, 0, stream>>>(vin,   Wv, bvp, Vt, 2, NKV, 256,    1.0f);

    // RPE MLPs
    rpe_kernel<<<960, 256, 0, stream>>>(ref2d, c1w1, c1b1, c1w2, c2w1, c2b1, c2w2,
                                        rpeX, rpeY);

    // Flash attention: 2*8*57 = 912 waves -> 228 blocks
    attn_kernel<<<228, 128, 0, stream>>>(Qh, Kh, Vt, rpeX, rpeY, pmask, Xh);

    // Output projection: 114*16 = 1824 waves -> 456 blocks
    outproj_kernel<<<456, 128, 0, stream>>>(Xh, Wp, bp, (float*)d_out);
}
